// MaskedRNNBase_30039001268625
// MI455X (gfx1250) — compile-verified, matrix operand order checked by reading, not run on hardware
//
#include <hip/hip_runtime.h>
#include <hip/hip_bf16.h>

// Problem constants (from reference): T=512, B=64, I=512, H=512, L=2 bidirectional
#define T_    512
#define B_    64
#define I_    512
#define H_    512
#define G_    2048   // 4*H
#define TWOH_ 1024   // 2*H
#define NBLK_ 64     // persistent grid: 2 dirs x 32 hidden tiles
#define KPAD_ 8      // bf16 elements of row padding -> LDS bank-conflict-free
#define KROWMAX_ (TWOH_ + H_ + KPAD_)   // 1544 bf16 per LDS weight row (layer 1)

typedef __attribute__((ext_vector_type(16))) __bf16 v16bf;
typedef __attribute__((ext_vector_type(8)))  float  v8f;

// Load one lane's half of a 16x32 bf16 WMMA tile: 8 contiguous bf16 at p and 8
// at p+16 (documented 16-bit A layout; B assumed symmetric with lane = column).
// Works for both global and LDS pointers (template preserves address space info
// through inlining -> ds_load_b128 for the LDS weight reads).
template <typename P>
static __device__ __forceinline__ v16bf load_half_tile(P p) {
  union { uint4 u[2]; v16bf v; } r;
  r.u[0] = *reinterpret_cast<const uint4*>(p);
  r.u[1] = *reinterpret_cast<const uint4*>(p + 16);
  return r.v;
}

static __device__ __forceinline__ float sigmoidf_(float x) {
  return 1.0f / (1.0f + __expf(-x));
}

__global__ void cvt_f32_bf16_kernel(const float* __restrict__ s,
                                    __bf16* __restrict__ d, int n) {
  int i = blockIdx.x * blockDim.x + threadIdx.x;
  if (i < n) d[i] = (__bf16)s[i];
}

__global__ void bias_sum_kernel(const float* __restrict__ a,
                                const float* __restrict__ b,
                                float* __restrict__ o, int n) {
  int i = blockIdx.x * blockDim.x + threadIdx.x;
  if (i < n) o[i] = a[i] + b[i];
}

__global__ void zero_u32_kernel(unsigned* p) { *p = 0u; }

// Device-wide split barrier: monotonic arrival counter in device memory.
// All NBLK_ blocks call it the same number of times; target = NBLK_ * epoch.
static __device__ __forceinline__ void grid_barrier(unsigned* cnt, unsigned target) {
  __syncthreads();
  if (threadIdx.x == 0) {
    __threadfence();  // release: make h/inter stores visible device-scope
    __hip_atomic_fetch_add(cnt, 1u, __ATOMIC_RELEASE, __HIP_MEMORY_SCOPE_AGENT);
    while (__hip_atomic_load(cnt, __ATOMIC_ACQUIRE, __HIP_MEMORY_SCOPE_AGENT) < target)
      __builtin_amdgcn_s_sleep(2);
  }
  __syncthreads();
  __threadfence();    // acquire: subsequent non-atomic h/x reads see fresh data
}

// Persistent bidirectional-LSTM kernel: 64 blocks x 128 threads (4 waves).
// Block -> (dir, 16-col hidden tile jt); wave -> 16-row batch tile mt.
// Per layer: stage the block's 4-gate weight slice (4x16 rows x (in+H) cols,
// bf16, +8 pad) into LDS once (128/192 KB of the 320 KB CDNA5 WGP LDS), then
// run all 512 timesteps with one grid barrier per step. Weights never leave
// LDS; only the 128 KB bf16 h state and x_t rows cross L2 per step.
__global__ __launch_bounds__(128)
void lstm_persistent_kernel(
    const __bf16* __restrict__ xbf,   // (T,B,I) layer-0 input, bf16
    __bf16* __restrict__ inter,       // (T,B,2H) layer-0 out / layer-1 in, bf16
    const __bf16* __restrict__ wih0f, const __bf16* __restrict__ whh0f,
    const __bf16* __restrict__ wih0r, const __bf16* __restrict__ whh0r,
    const __bf16* __restrict__ wih1f, const __bf16* __restrict__ whh1f,
    const __bf16* __restrict__ wih1r, const __bf16* __restrict__ whh1r,
    const float* __restrict__ bias0f, const float* __restrict__ bias0r,
    const float* __restrict__ bias1f, const float* __restrict__ bias1r,
    float* __restrict__ hS, float* __restrict__ cS,   // f32 states [2][B*H]
    __bf16* __restrict__ hbA, __bf16* __restrict__ hbB, // bf16 h ping-pong [2][B*H]
    const float* __restrict__ mask,   // (T,B) f32
    float* __restrict__ out,          // (T,B,2H) f32 final output
    float* __restrict__ hn_out, float* __restrict__ cn_out, // (4,B,H) each
    unsigned* __restrict__ bar)
{
  __shared__ alignas(16) __bf16 wlds[(size_t)64 * KROWMAX_];

  const int tid  = threadIdx.x;
  const int lane = tid & 31;
  const int mt   = tid >> 5;          // batch tile within block
  const int dir  = blockIdx.x >> 5;   // 0 = forward, 1 = reverse
  const int jt   = blockIdx.x & 31;   // hidden-column tile
  const int lm   = lane & 15;
  const int lh   = lane >> 4;
  const int m0   = mt * 16;
  const int j0   = jt * 16;
  const int ks   = 8 * lh;
  const int hj   = j0 + lm;
  const int BH   = B_ * H_;

  float* h_st = hS + (size_t)dir * BH;
  float* c_st = cS + (size_t)dir * BH;

  unsigned epoch = 0;

  for (int l = 0; l < 2; ++l) {
    const int in_dim = l ? TWOH_ : I_;
    const __bf16* xin = l ? inter : xbf;
    const __bf16* wih = l ? (dir ? wih1r : wih1f) : (dir ? wih0r : wih0f);
    const __bf16* whh = l ? (dir ? whh1r : whh1f) : (dir ? whh0r : whh0f);
    const float* bias = l ? (dir ? bias1r : bias1f) : (dir ? bias0r : bias0f);
    const int ktot = in_dim + H_;
    const int kp   = ktot + KPAD_;    // padded row stride: 4 dwords mod 64 banks
    const int cpr  = ktot >> 3;       // 16-byte chunks per LDS weight row

    // --- Stage this block's weight slice into LDS (row = gate*16 + col) ---
    for (int idx = tid; idx < 64 * cpr; idx += 128) {
      int row = idx / cpr;
      int kc  = (idx - row * cpr) << 3;
      int g = row >> 4, n = row & 15;
      size_t wrow = (size_t)(g * H_ + j0 + n);
      uint4 v;
      if (kc < in_dim) v = *reinterpret_cast<const uint4*>(wih + wrow * in_dim + kc);
      else             v = *reinterpret_cast<const uint4*>(whh + wrow * H_ + (kc - in_dim));
      *reinterpret_cast<uint4*>(&wlds[(size_t)row * kp + kc]) = v;
    }

    // --- Zero this wave's owned state elements (h0 = c0 = 0) ---
#pragma unroll
    for (int r = 0; r < 8; ++r) {
      int bm = m0 + r + 8 * lh;
      size_t si = (size_t)bm * H_ + hj;
      h_st[si] = 0.0f;
      c_st[si] = 0.0f;
      hbA[(size_t)dir * BH + si] = (__bf16)0.0f;
    }

    float bv[4];
#pragma unroll
    for (int g = 0; g < 4; ++g) bv[g] = bias[g * H_ + j0 + lm];

    grid_barrier(bar, (unsigned)NBLK_ * (++epoch));  // LDS staged + states zeroed

    // --- Time recurrence ---
    for (int t = 0; t < T_; ++t) {
      const int t_eff = dir ? (T_ - 1 - t) : t;
      const __bf16* h_rd = ((t & 1) ? hbB : hbA) + (size_t)dir * BH;
      __bf16*       h_wr = ((t & 1) ? hbA : hbB) + (size_t)dir * BH;

      v8f acc[4];
#pragma unroll
      for (int g = 0; g < 4; ++g)
#pragma unroll
        for (int r = 0; r < 8; ++r) acc[g][r] = bv[g];

      // x_t @ Wih^T (A from global/L2, B from LDS)
      const __bf16* arow = xin + ((size_t)t_eff * B_ + (m0 + lm)) * in_dim;
      for (int k0 = 0; k0 < in_dim; k0 += 32) {
        v16bf a = load_half_tile(arow + k0 + ks);
#pragma unroll
        for (int g = 0; g < 4; ++g) {
          v16bf b = load_half_tile(&wlds[(size_t)(g * 16 + lm) * kp + k0 + ks]);
          acc[g] = __builtin_amdgcn_wmma_f32_16x16x32_bf16(
              false, a, false, b, (short)0, acc[g], false, false);
        }
      }
      // h_{t-1} @ Whh^T
      const __bf16* hrow = h_rd + (size_t)(m0 + lm) * H_;
      for (int k0 = 0; k0 < H_; k0 += 32) {
        v16bf a = load_half_tile(hrow + k0 + ks);
#pragma unroll
        for (int g = 0; g < 4; ++g) {
          v16bf b = load_half_tile(&wlds[(size_t)(g * 16 + lm) * kp + in_dim + k0 + ks]);
          acc[g] = __builtin_amdgcn_wmma_f32_16x16x32_bf16(
              false, a, false, b, (short)0, acc[g], false, false);
        }
      }

      // Cell update + masking from C/D layout: M = r + 8*(lane>=16), N = lane&15
#pragma unroll
      for (int r = 0; r < 8; ++r) {
        int bm = m0 + r + 8 * lh;
        float mi = mask[(size_t)t_eff * B_ + bm];
        size_t si = (size_t)bm * H_ + hj;
        float cp = c_st[si];
        float hp = h_st[si];
        float ig = sigmoidf_(acc[0][r]);
        float fg = sigmoidf_(acc[1][r]);
        float gg = tanhf(acc[2][r]);
        float og = sigmoidf_(acc[3][r]);
        float c2 = fg * cp + ig * gg;
        float h2 = og * tanhf(c2);
        float cn = cp + (c2 - cp) * mi;
        float hn = hp + (h2 - hp) * mi;
        c_st[si] = cn;
        h_st[si] = hn;
        h_wr[si] = (__bf16)hn;
        size_t oi = ((size_t)t_eff * B_ + bm) * TWOH_ + (size_t)dir * H_ + hj;
        if (l == 0) inter[oi] = (__bf16)hn;
        else        out[oi]   = hn;
      }

      grid_barrier(bar, (unsigned)NBLK_ * (++epoch));  // h(t) visible to all
    }

    // --- Final h_n / c_n for this layer (masked scan => states == hT/cT) ---
#pragma unroll
    for (int r = 0; r < 8; ++r) {
      int bm = m0 + r + 8 * lh;
      size_t si = (size_t)bm * H_ + hj;
      size_t di = (size_t)(2 * l + dir) * BH + si;
      hn_out[di] = h_st[si];
      cn_out[di] = c_st[si];
    }
  }
}

extern "C" void kernel_launch(void* const* d_in, const int* in_sizes, int n_in,
                              void* d_out, int out_size, void* d_ws, size_t ws_size,
                              hipStream_t stream) {
  (void)in_sizes; (void)n_in; (void)out_size; (void)ws_size;
  const float* x    = (const float*)d_in[0];
  const float* mask = (const float*)d_in[1];
  // params: 4 groups (l0f, l0r, l1f, l1r) of (Wih, Whh, bih, bhh)
  const float* Wih[4]; const float* Whh[4]; const float* bih[4]; const float* bhh[4];
  for (int i = 0; i < 4; ++i) {
    Wih[i] = (const float*)d_in[2 + 4 * i + 0];
    Whh[i] = (const float*)d_in[2 + 4 * i + 1];
    bih[i] = (const float*)d_in[2 + 4 * i + 2];
    bhh[i] = (const float*)d_in[2 + 4 * i + 3];
  }

  // Workspace carve (~117 MB)
  char* p = (char*)d_ws;
  auto carve = [&](size_t bytes) -> void* {
    void* r = (void*)p;
    p += (bytes + 255) & ~(size_t)255;
    return r;
  };
  __bf16* xbf   = (__bf16*)carve((size_t)T_ * B_ * I_ * 2);
  __bf16* inter = (__bf16*)carve((size_t)T_ * B_ * TWOH_ * 2);
  __bf16* wihb[4]; __bf16* whhb[4]; float* biasw[4];
  for (int i = 0; i < 4; ++i) {
    int ind = (i < 2) ? I_ : TWOH_;
    wihb[i]  = (__bf16*)carve((size_t)G_ * ind * 2);
    whhb[i]  = (__bf16*)carve((size_t)G_ * H_ * 2);
    biasw[i] = (float*)carve((size_t)G_ * 4);
  }
  float*    hS  = (float*)carve((size_t)2 * B_ * H_ * 4);
  float*    cS  = (float*)carve((size_t)2 * B_ * H_ * 4);
  __bf16*   hbA = (__bf16*)carve((size_t)2 * B_ * H_ * 2);
  __bf16*   hbB = (__bf16*)carve((size_t)2 * B_ * H_ * 2);
  unsigned* bar = (unsigned*)carve(256);

  auto cvt = [&](const float* s, __bf16* d, int n) {
    cvt_f32_bf16_kernel<<<(n + 255) / 256, 256, 0, stream>>>(s, d, n);
  };

  // One-time conversions; all bf16 weights (~20 MB) stay L2-resident.
  cvt(x, xbf, T_ * B_ * I_);
  for (int i = 0; i < 4; ++i) {
    int ind = (i < 2) ? I_ : TWOH_;
    cvt(Wih[i], wihb[i], G_ * ind);
    cvt(Whh[i], whhb[i], G_ * H_);
    bias_sum_kernel<<<(G_ + 255) / 256, 256, 0, stream>>>(bih[i], bhh[i], biasw[i], G_);
  }

  float* out = (float*)d_out;
  const size_t OUT_ELEMS = (size_t)T_ * B_ * TWOH_;
  float* hn_out = out + OUT_ELEMS;
  float* cn_out = hn_out + (size_t)4 * B_ * H_;

  zero_u32_kernel<<<1, 1, 0, stream>>>(bar);  // re-armed every call (graph-safe)

  lstm_persistent_kernel<<<NBLK_, 128, 0, stream>>>(
      xbf, inter,
      wihb[0], whhb[0], wihb[1], whhb[1],
      wihb[2], whhb[2], wihb[3], whhb[3],
      biasw[0], biasw[1], biasw[2], biasw[3],
      hS, cS, hbA, hbB, mask,
      out, hn_out, cn_out, bar);
}
